// UnsupervisedHeteroSAGEModel_80453327389329
// MI455X (gfx1250) — compile-verified, matrix operand order checked by reading, not run on hardware
//
#include <hip/hip_runtime.h>

// ---------------------------------------------------------------------------
// Hetero GraphSAGE for MI455X (gfx1250, wave32, WMMA).
//   xA = x_A @ linA_W.T + b + emb_A[nidA];  xB = emb_B[nidB]
//   2 layers: segment-mean over edges + fused dual-GEMM + bias + ReLU.
// GEMMs: v_wmma_f32_16x16x32_bf16, f32 accumulate. Weights pre-packed into
// the WMMA B lane layout (bf16) once per launch; activations staged via LDS.
// Edge scatter: global f32 atomics (HBM-bound phase). Degrees hoisted
// (layer-invariant).
// ---------------------------------------------------------------------------

typedef __attribute__((ext_vector_type(16))) __bf16 v16bf;
typedef __attribute__((ext_vector_type(8)))  float  v8f;

#define HID 128
#define DFEAT 64
#define RT 5            // row tiles (of 16) per block: 80 rows, 6250/5 = 1250

// A-operand loader from an LDS-resident bf16 row (16x32 tile, wave32 layout):
// lane<16  : row m=lane,    K = kt+{0..7} and kt+{16..23}
// lane>=16 : row m=lane-16, K = kt+{8..15} and kt+{24..31}
__device__ __forceinline__ v16bf load_a_lds(const __bf16* __restrict__ row,
                                            int kt, int lane) {
  int khalf = (lane & 16) >> 1;           // 0 or 8
  const __bf16* p = row + kt + khalf;
  v16bf v;
#pragma unroll
  for (int i = 0; i < 8; ++i) { v[i] = p[i]; v[i + 8] = p[i + 16]; }
  return v;
}

// ---------------------------------------------------------------------------
// Pre-pack a row-major [128][K] f32 weight matrix into the WMMA B-operand
// lane layout as bf16: entry idx = (colTile*(K/32) + kt32)*32 + lane holds
// the v16bf that lane feeds to v_wmma for that (colTile, ktile).
// ---------------------------------------------------------------------------
__global__ void pack_w_kernel(const float* __restrict__ W,
                              __bf16* __restrict__ out, int K) {
  int idx = blockIdx.x * blockDim.x + threadIdx.x;
  const int KT = K >> 5;
  if (idx >= 8 * KT * 32) return;
  const int lane = idx & 31;
  const int kt32 = (idx >> 5) % KT;
  const int colTile = (idx >> 5) / KT;
  const int nl = lane & 15, khalf = (lane & 16) >> 1;
  const float* p = W + (size_t)(colTile * 16 + nl) * K + kt32 * 32 + khalf;
  v16bf v;
#pragma unroll
  for (int i = 0; i < 8; ++i) {
    v[i]     = (__bf16)p[i];
    v[i + 8] = (__bf16)p[i + 16];
  }
  ((v16bf*)out)[idx] = v;
}

// ---------------------------------------------------------------------------
// xA = x_A @ linA_W.T + linA_b + emb_A[nidA]   (M=N_A, K=64, N=128)
// ---------------------------------------------------------------------------
__global__ void __launch_bounds__(256)
fuse_input_kernel(const float* __restrict__ x, const int* __restrict__ nid,
                  const __bf16* __restrict__ WP, const float* __restrict__ b,
                  const float* __restrict__ emb, float* __restrict__ out) {
  __shared__ __bf16 sA[RT * 16 * DFEAT];
  const int rowBase = blockIdx.x * (RT * 16);

  for (int i = threadIdx.x; i < RT * 16 * DFEAT; i += 256)
    sA[i] = (__bf16)x[(size_t)rowBase * DFEAT + i];
  __syncthreads();

  const int wave = threadIdx.x >> 5, lane = threadIdx.x & 31;
  const int nl = lane & 15;
  const v16bf* wp = (const v16bf*)WP + (size_t)(wave * (DFEAT / 32)) * 32 + lane;

  v8f acc[RT] = {};
#pragma unroll
  for (int kt32 = 0; kt32 < DFEAT / 32; ++kt32) {
    v16bf b0 = wp[kt32 * 32];
#pragma unroll
    for (int rt = 0; rt < RT; ++rt) {
      v16bf a = load_a_lds(sA + (rt * 16 + nl) * DFEAT, kt32 * 32, lane);
      acc[rt] = __builtin_amdgcn_wmma_f32_16x16x32_bf16(false, a, false, b0,
                                                        (short)0, acc[rt],
                                                        false, false);
    }
  }

  const int n = wave * 16 + nl;
  const float bias = b[n];
#pragma unroll
  for (int rt = 0; rt < RT; ++rt) {
    const int rbase = rowBase + rt * 16 + ((lane & 16) >> 1);
#pragma unroll
    for (int r = 0; r < 8; ++r) {
      const int row = rbase + r;
      out[(size_t)row * HID + n] =
          acc[rt][r] + bias + emb[(size_t)nid[row] * HID + n];
    }
  }
}

// xB = emb_B[nidB]
__global__ void gather_copy_kernel(const float* __restrict__ emb,
                                   const int* __restrict__ nid,
                                   float* __restrict__ out, int nNodes) {
  int i = blockIdx.x * blockDim.x + threadIdx.x;
  if (i >= nNodes * HID) return;
  int node = i / HID, h = i - node * HID;
  out[i] = emb[(size_t)nid[node] * HID + h];
}

__global__ void zero4_kernel(float4* __restrict__ p, long long n4) {
  long long i = (long long)blockIdx.x * blockDim.x + threadIdx.x;
  if (i < n4) p[i] = make_float4(0.f, 0.f, 0.f, 0.f);
}

// Degrees are layer-invariant: deg[dst] += 1 per edge (computed once).
__global__ void degree_kernel(const int* __restrict__ edges, int E,
                              float* __restrict__ deg) {
  int e = blockIdx.x * blockDim.x + threadIdx.x;
  if (e < E) atomicAdd(deg + edges[E + e], 1.0f);
}

// ---------------------------------------------------------------------------
// Edge scatter: one wave per edge; lane l handles features l, l+32, l+64, l+96.
// sums[dst] += x[src]   (global f32 atomics, HBM-bound)
// ---------------------------------------------------------------------------
__global__ void __launch_bounds__(256)
scatter_kernel(const float* __restrict__ x, const int* __restrict__ edges,
               int E, float* __restrict__ sums) {
  int e = blockIdx.x * 8 + (threadIdx.x >> 5);
  if (e >= E) return;
  const int lane = threadIdx.x & 31;
  const int s = edges[e];
  const int d = edges[E + e];
  const float* xr = x + (size_t)s * HID;
  float* sr = sums + (size_t)d * HID;
#pragma unroll
  for (int i = 0; i < 4; ++i)
    atomicAdd(sr + lane + 32 * i, xr[lane + 32 * i]);
}

// ---------------------------------------------------------------------------
// new = relu( (sums/max(deg,1)) @ Wl.T + bl + xOld @ Wr.T )
// M=nodes, K=128, N=128. WlP/WrP pre-packed bf16 (B lane layout).
// out may alias sums (all sums reads complete before the barrier).
// ---------------------------------------------------------------------------
__global__ void __launch_bounds__(256)
sage_layer_kernel(const float* __restrict__ sums, const float* __restrict__ deg,
                  const float* __restrict__ xOld,
                  const __bf16* __restrict__ WlP, const float* __restrict__ bl,
                  const __bf16* __restrict__ WrP, float* __restrict__ out) {
  __shared__ __bf16 sM[RT * 16 * HID];   // mean messages (scaled), bf16
  __shared__ __bf16 sX[RT * 16 * HID];   // self features, bf16
  __shared__ float  sInv[RT * 16];
  const int rowBase = blockIdx.x * (RT * 16);

  for (int i = threadIdx.x; i < RT * 16; i += 256)
    sInv[i] = 1.0f / fmaxf(deg[rowBase + i], 1.0f);
  __syncthreads();

  for (int i = threadIdx.x; i < RT * 16 * HID; i += 256) {
    const int m = i >> 7;
    sM[i] = (__bf16)(sums[(size_t)rowBase * HID + i] * sInv[m]);
    sX[i] = (__bf16)xOld[(size_t)rowBase * HID + i];
  }
  __syncthreads();   // all sums reads complete -> in-place write is safe

  const int wave = threadIdx.x >> 5, lane = threadIdx.x & 31;
  const int nl = lane & 15;
  const v16bf* wl = (const v16bf*)WlP + (size_t)(wave * (HID / 32)) * 32 + lane;
  const v16bf* wr = (const v16bf*)WrP + (size_t)(wave * (HID / 32)) * 32 + lane;

  v8f acc[RT] = {};
#pragma unroll
  for (int kt32 = 0; kt32 < HID / 32; ++kt32) {
    v16bf b0 = wl[kt32 * 32];
    v16bf b1 = wr[kt32 * 32];
#pragma unroll
    for (int rt = 0; rt < RT; ++rt) {
      v16bf a0 = load_a_lds(sM + (rt * 16 + nl) * HID, kt32 * 32, lane);
      acc[rt] = __builtin_amdgcn_wmma_f32_16x16x32_bf16(false, a0, false, b0,
                                                        (short)0, acc[rt],
                                                        false, false);
      v16bf a1 = load_a_lds(sX + (rt * 16 + nl) * HID, kt32 * 32, lane);
      acc[rt] = __builtin_amdgcn_wmma_f32_16x16x32_bf16(false, a1, false, b1,
                                                        (short)0, acc[rt],
                                                        false, false);
    }
  }

  const int n = wave * 16 + nl;
  const float bias = bl[n];
#pragma unroll
  for (int rt = 0; rt < RT; ++rt) {
    const int rbase = rowBase + rt * 16 + ((lane & 16) >> 1);
#pragma unroll
    for (int r = 0; r < 8; ++r) {
      float v = acc[rt][r] + bias;
      out[(size_t)(rbase + r) * HID + n] = v > 0.0f ? v : 0.0f;
    }
  }
}

// ---------------------------------------------------------------------------
extern "C" void kernel_launch(void* const* d_in, const int* in_sizes, int n_in,
                              void* d_out, int out_size, void* d_ws, size_t ws_size,
                              hipStream_t stream) {
  const float* x_A  = (const float*)d_in[0];
  const int*   nidA = (const int*)d_in[1];
  const int*   nidB = (const int*)d_in[2];
  const int*   eAB  = (const int*)d_in[3];
  const int*   eBA  = (const int*)d_in[4];
  const float* embA = (const float*)d_in[5];
  const float* embB = (const float*)d_in[6];
  const float* linW = (const float*)d_in[7];
  const float* linb = (const float*)d_in[8];
  const float* WlAB = (const float*)d_in[9];
  const float* blAB = (const float*)d_in[10];
  const float* WrAB = (const float*)d_in[11];
  const float* WlBA = (const float*)d_in[12];
  const float* blBA = (const float*)d_in[13];
  const float* WrBA = (const float*)d_in[14];

  const int NA = 100000, NB = 100000, E = 800000;
  const size_t big = (size_t)NA * HID;       // == NB * HID
  float* P0A  = (float*)d_ws;
  float* P0B  = P0A + big;
  float* P1A  = P0B + big;
  float* P1B  = P1A + big;
  float* degA = P1B + big;
  float* degB = degA + NA;

  // packed bf16 weights (B lane layout)
  __bf16* linP = (__bf16*)(degB + NB);
  const size_t LIN_SZ = (size_t)HID * DFEAT;   // 8192 bf16
  const size_t W_SZ   = (size_t)HID * HID;     // 16384 bf16
  __bf16* pWlAB0 = linP + LIN_SZ;
  __bf16* pWrAB0 = pWlAB0 + W_SZ;
  __bf16* pWlBA0 = pWrAB0 + W_SZ;
  __bf16* pWrBA0 = pWlBA0 + W_SZ;
  __bf16* pWlAB1 = pWrBA0 + W_SZ;
  __bf16* pWrAB1 = pWlAB1 + W_SZ;
  __bf16* pWlBA1 = pWrAB1 + W_SZ;
  __bf16* pWrBA1 = pWlBA1 + W_SZ;

  const dim3 blk(256);
  const int gemmBlocks = NA / (RT * 16);      // 1250, exact
  const size_t WW = (size_t)HID * HID;        // f32 elements per weight matrix

  // ---- pack weights (once per launch; ~270 KB total) ----
  pack_w_kernel<<<2, blk, 0, stream>>>(linW, linP, DFEAT);
  pack_w_kernel<<<4, blk, 0, stream>>>(WlAB,      pWlAB0, HID);
  pack_w_kernel<<<4, blk, 0, stream>>>(WrAB,      pWrAB0, HID);
  pack_w_kernel<<<4, blk, 0, stream>>>(WlBA,      pWlBA0, HID);
  pack_w_kernel<<<4, blk, 0, stream>>>(WrBA,      pWrBA0, HID);
  pack_w_kernel<<<4, blk, 0, stream>>>(WlAB + WW, pWlAB1, HID);
  pack_w_kernel<<<4, blk, 0, stream>>>(WrAB + WW, pWrAB1, HID);
  pack_w_kernel<<<4, blk, 0, stream>>>(WlBA + WW, pWlBA1, HID);
  pack_w_kernel<<<4, blk, 0, stream>>>(WrBA + WW, pWrBA1, HID);

  // ---- input fusion ----
  fuse_input_kernel<<<gemmBlocks, blk, 0, stream>>>(x_A, nidA, linP, linb, embA, P0A);
  gather_copy_kernel<<<(NB * HID + 255) / 256, blk, 0, stream>>>(embB, nidB, P0B, NB);

  // ---- degrees (layer-invariant, compute once) ----
  zero4_kernel<<<(NA / 4 + 255) / 256, blk, 0, stream>>>((float4*)degA, NA / 4);
  zero4_kernel<<<(NB / 4 + 255) / 256, blk, 0, stream>>>((float4*)degB, NB / 4);
  degree_kernel<<<(E + 255) / 256, blk, 0, stream>>>(eAB, E, degB);
  degree_kernel<<<(E + 255) / 256, blk, 0, stream>>>(eBA, E, degA);

  const long long big4 = (long long)(big / 4);
  const unsigned zb = (unsigned)((big4 + 255) / 256);

  // ---- layer 0: x in P0*, sums/new in P1* ----
  zero4_kernel<<<zb, blk, 0, stream>>>((float4*)P1A, big4);
  zero4_kernel<<<zb, blk, 0, stream>>>((float4*)P1B, big4);
  scatter_kernel<<<(E + 7) / 8, blk, 0, stream>>>(P0A, eAB, E, P1B); // A->B
  scatter_kernel<<<(E + 7) / 8, blk, 0, stream>>>(P0B, eBA, E, P1A); // B->A
  sage_layer_kernel<<<gemmBlocks, blk, 0, stream>>>(P1B, degB, P0B,
      pWlAB0, blAB, pWrAB0, P1B);                   // newB in-place
  sage_layer_kernel<<<gemmBlocks, blk, 0, stream>>>(P1A, degA, P0A,
      pWlBA0, blBA, pWrBA0, P1A);                   // newA in-place

  // ---- layer 1: x in P1*, sums in P0*, outputs -> d_out = [xA ; xB] ----
  float* outA = (float*)d_out;
  float* outB = (float*)d_out + big;
  zero4_kernel<<<zb, blk, 0, stream>>>((float4*)P0A, big4);
  zero4_kernel<<<zb, blk, 0, stream>>>((float4*)P0B, big4);
  scatter_kernel<<<(E + 7) / 8, blk, 0, stream>>>(P1A, eAB, E, P0B);
  scatter_kernel<<<(E + 7) / 8, blk, 0, stream>>>(P1B, eBA, E, P0A);
  sage_layer_kernel<<<gemmBlocks, blk, 0, stream>>>(P0B, degB, P1B,
      pWlAB1, blAB + HID, pWrAB1, outB);
  sage_layer_kernel<<<gemmBlocks, blk, 0, stream>>>(P0A, degA, P1A,
      pWlBA1, blBA + HID, pWrBA1, outA);
}
